// GATEncoder_88742614270162
// MI455X (gfx1250) — matrix operand order repaired and checked by reading, not compiled, not run on hardware
//
#include <hip/hip_runtime.h>
#include <hip/hip_bf16.h>

// ---------------------------------------------------------------------------
// Types for CDNA5 WMMA
// ---------------------------------------------------------------------------
typedef __attribute__((ext_vector_type(16))) __bf16 v16bf;
typedef __attribute__((ext_vector_type(2)))  __bf16 v2bf;
typedef __attribute__((ext_vector_type(8)))  float  v8f;

#define TM 64
#define TN 64
#define TK 64   // two 16x16x32 WMMA K-substeps per staged tile

#define EPI_NONE   0
#define EPI_RELU   1
#define EPI_SIGACC 2

__device__ __forceinline__ unsigned short f2bf(float f) {
  unsigned u = __float_as_uint(f);
  unsigned r = (u >> 16) & 1u;          // round-to-nearest-even
  u += 0x7fffu + r;
  return (unsigned short)(u >> 16);
}

__device__ __forceinline__ unsigned pack2bf(float lo, float hi) {
#if __has_builtin(__builtin_amdgcn_cvt_pk_bf16_f32)
  v2bf p = __builtin_amdgcn_cvt_pk_bf16_f32(lo, hi);
  return *(unsigned*)&p;                // v_cvt_pk_bf16_f32: 1 VALU op
#else
  return ((unsigned)f2bf(hi) << 16) | (unsigned)f2bf(lo);
#endif
}

union FragBF { uint4 q[2]; v16bf v; };

// ---------------------------------------------------------------------------
// Tiled bf16 WMMA GEMM:  C[M,N] = epilogue(A[M,K] @ B[K,N] + bias)
//
// 256 threads = 8 waves; 64x64 C tile; each wave owns a 16x32 strip.
// A tile: row-major bf16 in LDS (stride 64). A fragment (k = 16*half + s, same order as B) is
//   32 contiguous bytes per lane -> 2x ds_load_b128.
// B tile: COLUMN-major bf16 in LDS (sB[n*64+k]); B fragment is 32 contiguous
//   bytes per lane -> 2x ds_load_b128. Transpose is folded into staging.
// Global->reg->LDS with register double-buffering: next tile's global loads
// are issued before the WMMA burst of the current tile.
// ---------------------------------------------------------------------------
__global__ __launch_bounds__(256)
void gemm_bf16_wmma(const float* __restrict__ A, const float* __restrict__ B,
                    const float* __restrict__ bias, float* __restrict__ C,
                    int M, int N, int K, int mode)
{
  __shared__ __align__(16) unsigned short sA[TM * TK];   // 64x64 bf16, 8KB
  __shared__ __align__(16) unsigned short sB[TN * TK];   // col-major 64x64, 8KB

  const int tid  = threadIdx.x;
  const int lane = tid & 31;
  const int wave = tid >> 5;
  const int wm   = (wave >> 1) * 16;       // wave M offset in tile: 0,16,32,48
  const int wn   = (wave & 1) * 32;        // wave N offset in tile: 0,32
  const int bM   = blockIdx.y * TM;
  const int bN   = blockIdx.x * TN;

  const int m16  = lane & 15;
  const int half = lane >> 4;

  // staging assignments
  const int ar = tid >> 2;              // A row in tile (0..63)
  const int ac = (tid & 3) * 16;        // A col start (0,16,32,48)
  const int bn = tid & 63;              // B column in tile (0..63)
  const int bk = (tid >> 6) * 16;       // B k start (0,16,32,48)

  const bool arow_ok = (bM + ar) < M;
  const float4* __restrict__ apv =
      (const float4*)(A + (size_t)(bM + ar) * K + ac);      // + kt/4 per tile
  const float*  __restrict__ bp = B + (size_t)bk * N + bN + bn; // + kt*N per tile

  float4 ra[4];                     // 16 floats of A (one row chunk)
  float  rb[16];                    // 16 floats of B (one column chunk)

  auto load_tile = [&](int kt) {
    if (arow_ok) {
      const float4* p = apv + (kt >> 2);
      ra[0] = p[0]; ra[1] = p[1]; ra[2] = p[2]; ra[3] = p[3];
    } else {
      ra[0] = ra[1] = ra[2] = ra[3] = make_float4(0.f, 0.f, 0.f, 0.f);
    }
    const float* q = bp + (size_t)kt * N;
    #pragma unroll
    for (int j = 0; j < 16; ++j) rb[j] = q[(size_t)j * N];
  };

  auto store_tile = [&]() {
    uint4 qa0, qa1;
    const float* a = (const float*)ra;
    qa0.x = pack2bf(a[0],  a[1]);  qa0.y = pack2bf(a[2],  a[3]);
    qa0.z = pack2bf(a[4],  a[5]);  qa0.w = pack2bf(a[6],  a[7]);
    qa1.x = pack2bf(a[8],  a[9]);  qa1.y = pack2bf(a[10], a[11]);
    qa1.z = pack2bf(a[12], a[13]); qa1.w = pack2bf(a[14], a[15]);
    uint4* da = (uint4*)&sA[ar * TK + ac];
    da[0] = qa0; da[1] = qa1;

    uint4 qb0, qb1;
    qb0.x = pack2bf(rb[0],  rb[1]);  qb0.y = pack2bf(rb[2],  rb[3]);
    qb0.z = pack2bf(rb[4],  rb[5]);  qb0.w = pack2bf(rb[6],  rb[7]);
    qb1.x = pack2bf(rb[8],  rb[9]);  qb1.y = pack2bf(rb[10], rb[11]);
    qb1.z = pack2bf(rb[12], rb[13]); qb1.w = pack2bf(rb[14], rb[15]);
    uint4* db = (uint4*)&sB[bn * TK + bk];
    db[0] = qb0; db[1] = qb1;
  };

  v8f acc0 = {};
  v8f acc1 = {};

  // prologue: stage tile 0
  load_tile(0);
  store_tile();

  for (int kt = 0;;) {
    __syncthreads();                       // LDS tile ready

    const int ktn = kt + TK;
    if (ktn < K) load_tile(ktn);           // prefetch next tile into registers

    // two WMMA K-substeps over the staged 64-wide tile
    #pragma unroll
    for (int s = 0; s < 2; ++s) {
      const int ks = s * 32;
      FragBF fa, fb0, fb1;
      {
        const int base = (wm + m16) * TK + ks + half * 16;
        fa.q[0] = *(const uint4*)&sA[base];
        fa.q[1] = *(const uint4*)&sA[base + 8];
      }
      {
        const int b0 = (wn + m16) * TK + ks + half * 16;
        fb0.q[0] = *(const uint4*)&sB[b0];
        fb0.q[1] = *(const uint4*)&sB[b0 + 8];
        const int b1 = (wn + 16 + m16) * TK + ks + half * 16;
        fb1.q[0] = *(const uint4*)&sB[b1];
        fb1.q[1] = *(const uint4*)&sB[b1 + 8];
      }
      acc0 = __builtin_amdgcn_wmma_f32_16x16x32_bf16(false, fa.v, false, fb0.v,
                                                     (short)0, acc0, false, false);
      acc1 = __builtin_amdgcn_wmma_f32_16x16x32_bf16(false, fa.v, false, fb1.v,
                                                     (short)0, acc1, false, false);
    }

    kt = ktn;
    if (kt >= K) break;
    __syncthreads();                       // all reads done before overwrite
    store_tile();
  }

  // Epilogue. C/D layout: lane-half selects M+8, VGPR index r = M row, lane&15 = N.
  #pragma unroll
  for (int r = 0; r < 8; ++r) {
    const int m    = wm + half * 8 + r;
    const int grow = bM + m;
    if (grow >= M) continue;
    const int n0 = bN + wn + m16;
    const int n1 = n0 + 16;
    float v0 = acc0[r] + (bias ? bias[n0] : 0.f);
    float v1 = acc1[r] + (bias ? bias[n1] : 0.f);
    float* c0 = C + (size_t)grow * N + n0;
    float* c1 = C + (size_t)grow * N + n1;
    if (mode == EPI_RELU) {
      *c0 = v0 > 0.f ? v0 : 0.f;
      *c1 = v1 > 0.f ? v1 : 0.f;
    } else if (mode == EPI_SIGACC) {
      *c0 += 1.f / (1.f + expf(-v0));
      *c1 += 1.f / (1.f + expf(-v1));
    } else {
      *c0 = v0;
      *c1 = v1;
    }
  }
}

// ---------------------------------------------------------------------------
// Per-row dual dot product: s1[r] = H[r,:].a1 ; s2[r] = H[r,:].a2
// ---------------------------------------------------------------------------
__global__ __launch_bounds__(256)
void rowdot2(const float* __restrict__ H, const float* __restrict__ a1,
             const float* __restrict__ a2, float* __restrict__ s1,
             float* __restrict__ s2, int M, int F)
{
  const int wave = threadIdx.x >> 5;
  const int lane = threadIdx.x & 31;
  const int row  = blockIdx.x * 8 + wave;
  if (row >= M) return;
  const float* h = H + (size_t)row * F;
  float x1 = 0.f, x2 = 0.f;
  for (int f = lane; f < F; f += 32) {
    const float v = h[f];
    x1 += v * a1[f];
    x2 += v * a2[f];
  }
  for (int o = 16; o; o >>= 1) {
    x1 += __shfl_xor(x1, o, 32);
    x2 += __shfl_xor(x2, o, 32);
  }
  if (lane == 0) { s1[row] = x1; s2[row] = x2; }
}

// ---------------------------------------------------------------------------
// Segment softmax over incoming edges (+self loops) via global atomics.
// ---------------------------------------------------------------------------
__device__ __forceinline__ unsigned f2ord(float f) {
  unsigned u = __float_as_uint(f);
  return (u & 0x80000000u) ? ~u : (u | 0x80000000u);
}
__device__ __forceinline__ float ord2f(unsigned u) {
  return (u & 0x80000000u) ? __uint_as_float(u & 0x7fffffffu)
                           : __uint_as_float(~u);
}
__device__ __forceinline__ float lrelu02(float v) { return v > 0.f ? v : 0.2f * v; }

__global__ void init_softmax(unsigned* __restrict__ mmax, float* __restrict__ den, int Nn)
{
  const int i = blockIdx.x * blockDim.x + threadIdx.x;
  if (i < Nn) { mmax[i] = f2ord(-__builtin_inff()); den[i] = 0.f; }
}

__global__ void zero_f32(float* __restrict__ p, size_t n)
{
  for (size_t i = (size_t)blockIdx.x * blockDim.x + threadIdx.x; i < n;
       i += (size_t)gridDim.x * blockDim.x)
    p[i] = 0.f;
}

__global__ void edge_max(const int* __restrict__ ei, int E, int Nn,
                         const float* __restrict__ ss, const float* __restrict__ sd,
                         unsigned* __restrict__ mmax)
{
  const int i = blockIdx.x * blockDim.x + threadIdx.x;
  if (i >= E + Nn) return;
  int s, d;
  if (i < E) { s = ei[i]; d = ei[E + i]; } else { s = d = i - E; }
  atomicMax(&mmax[d], f2ord(lrelu02(ss[s] + sd[d])));
}

__global__ void edge_sum(const int* __restrict__ ei, int E, int Nn,
                         const float* __restrict__ ss, const float* __restrict__ sd,
                         const unsigned* __restrict__ mmax, float* __restrict__ den)
{
  const int i = blockIdx.x * blockDim.x + threadIdx.x;
  if (i >= E + Nn) return;
  int s, d;
  if (i < E) { s = ei[i]; d = ei[E + i]; } else { s = d = i - E; }
  const float e = lrelu02(ss[s] + sd[d]);
  atomicAdd(&den[d], expf(e - ord2f(mmax[d])));
}

// One block per edge: out[dst,:] += coef * H[src,:]
__global__ __launch_bounds__(256)
void edge_scatter(const int* __restrict__ ei, int E, int Nn,
                  const float* __restrict__ ss, const float* __restrict__ sd,
                  const unsigned* __restrict__ mmax, const float* __restrict__ den,
                  const float* __restrict__ H, float* __restrict__ out, int F)
{
  const int i = blockIdx.x;
  int s, d;
  if (i < E) { s = ei[i]; d = ei[E + i]; } else { s = d = i - E; }
  const float e    = lrelu02(ss[s] + sd[d]);
  const float coef = expf(e - ord2f(mmax[d])) / den[d];
  const float* hs = H + (size_t)s * F;
  float*       od = out + (size_t)d * F;
  for (int f = threadIdx.x; f < F; f += blockDim.x)
    atomicAdd(&od[f], coef * hs[f]);
}

// ---------------------------------------------------------------------------
// Fused: v = agg + bias1 + sk + bias2 ; y = LeakyReLU(LayerNorm(v)*g + be)
// ---------------------------------------------------------------------------
__global__ __launch_bounds__(256)
void fuse_add_ln_lrelu(const float* __restrict__ agg, const float* __restrict__ bias1,
                       const float* __restrict__ sk,  const float* __restrict__ bias2,
                       const float* __restrict__ g,   const float* __restrict__ be,
                       float* __restrict__ out, int F)
{
  const int row = blockIdx.x;
  const float* pa = agg + (size_t)row * F;
  const float* ps = sk  + (size_t)row * F;
  float lsum = 0.f, lsq = 0.f;
  for (int f = threadIdx.x; f < F; f += blockDim.x) {
    const float v = pa[f] + bias1[f] + ps[f] + bias2[f];
    lsum += v; lsq += v * v;
  }
  for (int o = 16; o; o >>= 1) {
    lsum += __shfl_xor(lsum, o, 32);
    lsq  += __shfl_xor(lsq,  o, 32);
  }
  __shared__ float s1[8], s2[8];
  __shared__ float mu_s, rs_s;
  const int wave = threadIdx.x >> 5, lane = threadIdx.x & 31;
  if (lane == 0) { s1[wave] = lsum; s2[wave] = lsq; }
  __syncthreads();
  if (threadIdx.x == 0) {
    float a = 0.f, b = 0.f;
    #pragma unroll
    for (int i = 0; i < 8; ++i) { a += s1[i]; b += s2[i]; }
    const float mu  = a / (float)F;
    const float var = b / (float)F - mu * mu;
    mu_s = mu;
    rs_s = rsqrtf(var + 1e-5f);
  }
  __syncthreads();
  const float mu = mu_s, rs = rs_s;
  float* po = out + (size_t)row * F;
  for (int f = threadIdx.x; f < F; f += blockDim.x) {
    const float v = pa[f] + bias1[f] + ps[f] + bias2[f];
    const float y = (v - mu) * rs * g[f] + be[f];
    po[f] = y > 0.f ? y : 0.2f * y;
  }
}

// x *= attsum / 16
__global__ void gate_mul(float* __restrict__ x, const float* __restrict__ attsum, size_t n)
{
  for (size_t i = (size_t)blockIdx.x * blockDim.x + threadIdx.x; i < n;
       i += (size_t)gridDim.x * blockDim.x)
    x[i] *= attsum[i] * 0.0625f;
}

// ---------------------------------------------------------------------------
// Host-side helpers
// ---------------------------------------------------------------------------
static inline void launch_gemm(hipStream_t st, const float* A, const float* B,
                               const float* bias, float* C, int M, int N, int K, int mode)
{
  dim3 grid(N / TN, (M + TM - 1) / TM), blk(256);
  hipLaunchKernelGGL(gemm_bf16_wmma, grid, blk, 0, st, A, B, bias, C, M, N, K, mode);
}

static inline void run_gat(hipStream_t st, const int* ei, int E, int Nn,
                           const float* H, int F,
                           const float* av_src, const float* av_dst,
                           float* ssrc, float* sdst, unsigned* mmax, float* den,
                           float* out)
{
  hipLaunchKernelGGL(rowdot2, dim3((Nn + 7) / 8), dim3(256), 0, st,
                     H, av_src, av_dst, ssrc, sdst, Nn, F);
  hipLaunchKernelGGL(zero_f32, dim3(2048), dim3(256), 0, st, out, (size_t)Nn * F);
  hipLaunchKernelGGL(init_softmax, dim3((Nn + 255) / 256), dim3(256), 0, st, mmax, den, Nn);
  const int T = E + Nn;
  hipLaunchKernelGGL(edge_max, dim3((T + 255) / 256), dim3(256), 0, st,
                     ei, E, Nn, ssrc, sdst, mmax);
  hipLaunchKernelGGL(edge_sum, dim3((T + 255) / 256), dim3(256), 0, st,
                     ei, E, Nn, ssrc, sdst, mmax, den);
  hipLaunchKernelGGL(edge_scatter, dim3(T), dim3(256), 0, st,
                     ei, E, Nn, ssrc, sdst, mmax, den, H, out, F);
}

// ---------------------------------------------------------------------------
// kernel_launch: full 3-block GAT encoder pipeline
// ---------------------------------------------------------------------------
extern "C" void kernel_launch(void* const* d_in, const int* in_sizes, int n_in,
                              void* d_out, int out_size, void* d_ws, size_t ws_size,
                              hipStream_t stream)
{
  const float* x    = (const float*)d_in[0];
  const int*   ei   = (const int*)  d_in[1];
  const float* W1   = (const float*)d_in[2];
  const float* b1   = (const float*)d_in[3];
  const float* a1s  = (const float*)d_in[4];
  const float* a1d  = (const float*)d_in[5];
  const float* Wsk1 = (const float*)d_in[6];
  const float* bsk1 = (const float*)d_in[7];
  const float* g1   = (const float*)d_in[8];
  const float* be1  = (const float*)d_in[9];
  const float* Wh1  = (const float*)d_in[10];
  const float* bh1  = (const float*)d_in[11];
  const float* Wh2  = (const float*)d_in[12];
  const float* bh2  = (const float*)d_in[13];
  const float* Wsk2 = (const float*)d_in[14];
  const float* bsk2 = (const float*)d_in[15];
  const float* W2   = (const float*)d_in[16];
  const float* b2   = (const float*)d_in[17];
  const float* a2s  = (const float*)d_in[18];
  const float* a2d  = (const float*)d_in[19];
  const float* g2   = (const float*)d_in[20];
  const float* be2  = (const float*)d_in[21];
  const float* Wsk3 = (const float*)d_in[22];
  const float* bsk3 = (const float*)d_in[23];
  const float* W3   = (const float*)d_in[24];
  const float* b3   = (const float*)d_in[25];
  const float* a3s  = (const float*)d_in[26];
  const float* a3d  = (const float*)d_in[27];
  const float* g3   = (const float*)d_in[28];
  const float* be3  = (const float*)d_in[29];

  const int Nn = in_sizes[0] / 512;   // 10000
  const int E  = in_sizes[1] / 2;     // 80000

  const size_t big  = (size_t)Nn * 2048;
  const size_t need = (4 * big + 4 * (size_t)Nn) * sizeof(float);
  if (ws_size < need) return;

  float*    bufA = (float*)d_ws;        // h1 -> attsum -> h2 -> h3
  float*    bufB = bufA + big;          // sk1 -> hm -> sk2 -> sk3
  float*    bufC = bufB + big;          // GAT aggregation output
  float*    bufD = bufC + big;          // x1 -> x1*att -> x2
  float*    ssrc = bufD + big;
  float*    sdst = ssrc + Nn;
  unsigned* mmax = (unsigned*)(sdst + Nn);
  float*    den  = (float*)(mmax + Nn);
  float*    y    = (float*)d_out;       // final [Nn,512]

  // ---- Block 1: 512 -> 2048 ----
  launch_gemm(stream, x, W1,   nullptr, bufA, Nn, 2048, 512, EPI_NONE);   // h1
  launch_gemm(stream, x, Wsk1, nullptr, bufB, Nn, 2048, 512, EPI_NONE);   // skip
  run_gat(stream, ei, E, Nn, bufA, 2048, a1s, a1d, ssrc, sdst, mmax, den, bufC);
  hipLaunchKernelGGL(fuse_add_ln_lrelu, dim3(Nn), dim3(256), 0, stream,
                     bufC, b1, bufB, bsk1, g1, be1, bufD, 2048);          // x1

  // ---- 16-head gating ensemble (compute-dominant: ~670 GFLOP) ----
  hipLaunchKernelGGL(zero_f32, dim3(2048), dim3(256), 0, stream, bufA, big);
  for (int h = 0; h < 16; ++h) {
    const float* wh1 = Wh1 + (size_t)h * 2048 * 512;
    const float* wh2 = Wh2 + (size_t)h * 512 * 2048;
    launch_gemm(stream, bufD, wh1, bh1 + h * 512,  bufB, Nn, 512,  2048, EPI_RELU);
    launch_gemm(stream, bufB, wh2, bh2 + h * 2048, bufA, Nn, 2048, 512,  EPI_SIGACC);
  }
  hipLaunchKernelGGL(gate_mul, dim3(2048), dim3(256), 0, stream, bufD, bufA, big);

  // ---- Block 2: 2048 -> 1024 ----
  launch_gemm(stream, bufD, W2,   nullptr, bufA, Nn, 1024, 2048, EPI_NONE);
  launch_gemm(stream, bufD, Wsk2, nullptr, bufB, Nn, 1024, 2048, EPI_NONE);
  run_gat(stream, ei, E, Nn, bufA, 1024, a2s, a2d, ssrc, sdst, mmax, den, bufC);
  hipLaunchKernelGGL(fuse_add_ln_lrelu, dim3(Nn), dim3(256), 0, stream,
                     bufC, b2, bufB, bsk2, g2, be2, bufD, 1024);          // x2

  // ---- Block 3: 1024 -> 512 ----
  launch_gemm(stream, bufD, W3,   nullptr, bufA, Nn, 512, 1024, EPI_NONE);
  launch_gemm(stream, bufD, Wsk3, nullptr, bufB, Nn, 512, 1024, EPI_NONE);
  run_gat(stream, ei, E, Nn, bufA, 512, a3s, a3d, ssrc, sdst, mmax, den, bufC);
  hipLaunchKernelGGL(fuse_add_ln_lrelu, dim3(Nn), dim3(256), 0, stream,
                     bufC, b3, bufB, bsk3, g3, be3, y, 512);              // x3 -> d_out
}
